// CoreNetwork_47115791237988
// MI455X (gfx1250) — compile-verified, matrix operand order checked by reading, not run on hardware
//
#include <hip/hip_runtime.h>
#include <hip/hip_bf16.h>
#include <math.h>

#define B_  16
#define N_  50
#define E_  10
#define H_  64
#define F_  256
#define OUT_ 3

typedef __attribute__((ext_vector_type(16))) __bf16 bf16x16;
typedef __attribute__((ext_vector_type(8)))  float  v8f;

__device__ __forceinline__ unsigned short f2bf(float f) {
    unsigned int u = __builtin_bit_cast(unsigned int, f);
    unsigned int r = u + 0x7FFFu + ((u >> 16) & 1u);   // RNE
    return (unsigned short)(r >> 16);
}

// CDNA5 hardware tanh (V_TANH_F32, TRANS32 op). Branch-free: keeps EXEC
// untouched in the WMMA hot loop and co-executes with the matrix pipe.
__device__ __forceinline__ float fast_tanh(float x) {
#if __has_builtin(__builtin_amdgcn_tanhf)
    return __builtin_amdgcn_tanhf(x);
#else
    float y;
    // ISA 7.4: TRANS result needs 1 independent op / V_NOP before use.
    asm volatile("v_tanh_f32 %0, %1\n\tv_nop" : "=v"(y) : "v"(x));
    return y;
#endif
}

// ISA 7.12.2: 16-bit A (16x32) / B (32x16) fragment K index for
// (lane-group g = lane>>4, slot s = packed half index 0..15, vgpr v = s>>1)
__device__ __forceinline__ int kidx16(int g, int s) {
    int v = s >> 1, half = s & 1;
    int base = (v < 4) ? (2 * v + 8 * g) : (16 + 2 * (v - 4) + 8 * g);
    return base + half;
}

__device__ __forceinline__ float sigmoidf_(float x) {
    return 1.0f / (1.0f + __expf(-x));
}

// ---------------------------------------------------------------------------
// Kernel 1: e1[b,i,j,:] = sigmoid(edges[b,i,j,:] @ We1^T + be1), bf16,
// stored TRANSPOSED as e1T[b][j][i][128] for contiguous per-(b,j) reads.
// ---------------------------------------------------------------------------
__global__ __launch_bounds__(128) void edge_l1_kernel(
    const float* __restrict__ edges, const float* __restrict__ We1,
    const float* __restrict__ be1, unsigned short* __restrict__ e1T)
{
    const int r = blockIdx.x;                 // (b*N + i)*N + j
    const int t = threadIdx.x;                // 0..127 output unit
    const int b = r / (N_ * N_);
    const int rem = r % (N_ * N_);
    const int i = rem / N_;
    const int j = rem % N_;

    __shared__ float ev[E_];
    if (t < E_) ev[t] = edges[(size_t)r * E_ + t];
    __syncthreads();

    float a = be1[t];
    #pragma unroll
    for (int c = 0; c < E_; ++c) a += We1[t * E_ + c] * ev[c];
    const float sg = sigmoidf_(a);
    e1T[((size_t)(b * N_ + j) * N_ + i) * 128 + t] = f2bf(sg);
}

// ---------------------------------------------------------------------------
// Kernel 2: pack We2 [4096,128] fp32 into bf16 WMMA-B fragments.
// Tile (nt,kt): B(32x16) = We2T[kt*32.., nt*16..]; lane-contiguous 16 halves.
// ---------------------------------------------------------------------------
__global__ __launch_bounds__(32) void pack_we2_kernel(
    const float* __restrict__ We2, unsigned short* __restrict__ we2p)
{
    const int tile = blockIdx.x;              // 0..1023: nt = tile>>2, kt = tile&3
    const int lane = threadIdx.x;             // 0..31
    const int nt = tile >> 2, kt = tile & 3;
    const int g = lane >> 4, ln = lane & 15;
    const int n = nt * 16 + ln;               // output-unit (dk) index 0..4095
    #pragma unroll
    for (int s = 0; s < 16; ++s) {
        const int k = kt * 32 + kidx16(g, s); // inner-dim index 0..127
        we2p[(size_t)tile * 512 + lane * 16 + s] = f2bf(We2[(size_t)n * 128 + k]);
    }
}

// ---------------------------------------------------------------------------
// Kernel 3 (hot, WMMA): fused edge_wgt recompute + message contraction.
// One workgroup per (b,j). msgs[b,j,d] = (1/N^2) * sum_{i,k} tanh(
//     (e1[b,i,j,:] @ We2^T)[d*64+k] + be2[d*64+k]) * h[b,i,k]
// GEMM: A = e1 rows over i (50->64 padded) x K=128 (bf16),
//       B = We2^T (packed), C tiles tanh'ed and reduced on the fly.
// ---------------------------------------------------------------------------
__global__ __launch_bounds__(256) void fused_msgs_kernel(
    const unsigned short* __restrict__ e1T,
    const unsigned short* __restrict__ we2p,
    const float* __restrict__ be2,
    const float* __restrict__ h_in,
    float* __restrict__ msgs)
{
    __shared__ unsigned short e1s[64 * 128];  // 16 KB, rows 50..63 zero
    __shared__ float hs[64 * 64];             // 16 KB, rows 50..63 zero
    __shared__ float msgAcc[H_];

    const int bj = blockIdx.x;                // 0..799
    const int t  = threadIdx.x;

    const unsigned short* esrc = e1T + (size_t)bj * (N_ * 128);
    for (int idx = t; idx < 64 * 128; idx += 256)
        e1s[idx] = (idx < N_ * 128) ? esrc[idx] : (unsigned short)0;

    const int b = bj / N_;
    const float* hsrc = h_in + (size_t)b * (N_ * H_);
    for (int idx = t; idx < 64 * 64; idx += 256)
        hs[idx] = (idx < N_ * H_) ? hsrc[idx] : 0.0f;

    if (t < H_) msgAcc[t] = 0.0f;
    __syncthreads();

    const int wave = t >> 5;                  // 0..7
    const int lane = t & 31;
    const int g  = lane >> 4;
    const int ln = lane & 15;

    union AFrag { bf16x16 v; unsigned short u[16]; };

    for (int mt = 0; mt < 4; ++mt) {
        // Build A fragments (rows i = mt*16+ln, K-steps kt*32..) from LDS.
        AFrag af[4];
        const int i_a = mt * 16 + ln;
        #pragma unroll
        for (int kt = 0; kt < 4; ++kt)
            #pragma unroll
            for (int s = 0; s < 16; ++s)
                af[kt].u[s] = e1s[i_a * 128 + kt * 32 + kidx16(g, s)];

        for (int nt = wave; nt < 256; nt += 8) {
            const int col  = nt * 16 + ln;    // dk column 0..4095
            const float bias = be2[col];
            const int d = col >> 6;           // single d per 16-wide N tile
            const int k = col & 63;

            v8f c = {};
            #pragma unroll
            for (int kt = 0; kt < 4; ++kt) {
                union { bf16x16 v; uint4 q[2]; } bf;
                const uint4* p = (const uint4*)(we2p +
                    ((size_t)(nt * 4 + kt) * 512 + lane * 16));
                bf.q[0] = p[0];
                bf.q[1] = p[1];
                c = __builtin_amdgcn_wmma_f32_16x16x32_bf16(
                        false, af[kt].v, false, bf.v,
                        (short)0, c, false, false);
            }

            // Epilogue: hw tanh + multiply by h[i,k], reduce tile -> msgAcc[d].
            // C layout: vgpr r, lanes 0-15: M=r, N=lane; lanes 16-31: M=r+8.
            float acc = 0.0f;
            #pragma unroll
            for (int r = 0; r < 8; ++r) {
                const int i = mt * 16 + r + 8 * g;
                acc += fast_tanh(c[r] + bias) * hs[i * 64 + k];
            }
            #pragma unroll
            for (int off = 16; off > 0; off >>= 1)
                acc += __shfl_xor(acc, off, 32);
            if (lane == 0) atomicAdd(&msgAcc[d], acc);
        }
    }
    __syncthreads();
    if (t < H_)
        msgs[(size_t)bj * H_ + t] = msgAcc[t] * (1.0f / (N_ * N_));
}

// ---------------------------------------------------------------------------
// Kernel 4: PyTorch-semantics GRUCell + mask. One block per (b,n) row.
// ---------------------------------------------------------------------------
__global__ __launch_bounds__(64) void gru_kernel(
    const float* __restrict__ msgs, const float* __restrict__ h_in,
    const float* __restrict__ W_ih, const float* __restrict__ W_hh,
    const float* __restrict__ b_ih, const float* __restrict__ b_hh,
    const float* __restrict__ masks, float* __restrict__ h_out)
{
    const int m = blockIdx.x;                 // 0..799
    const int u = threadIdx.x;                // 0..63
    __shared__ float xs[H_], hh[H_];
    xs[u] = msgs[(size_t)m * H_ + u];
    hh[u] = h_in[(size_t)m * H_ + u];
    __syncthreads();

    float gi[3], gh[3];
    #pragma unroll
    for (int gate = 0; gate < 3; ++gate) {
        const int row = gate * H_ + u;
        float ai = b_ih[row], ah = b_hh[row];
        const float* wi = W_ih + (size_t)row * H_;
        const float* wh = W_hh + (size_t)row * H_;
        for (int k = 0; k < H_; ++k) {
            ai += wi[k] * xs[k];
            ah += wh[k] * hh[k];
        }
        gi[gate] = ai; gh[gate] = ah;
    }
    const float r = sigmoidf_(gi[0] + gh[0]);
    const float z = sigmoidf_(gi[1] + gh[1]);
    const float n = fast_tanh(gi[2] + r * gh[2]);
    const float hn = (1.0f - z) * n + z * hh[u];
    h_out[(size_t)m * H_ + u] = hn * masks[m];
}

// ---------------------------------------------------------------------------
// Kernel 5: LatentNN head. One block per (b,n) row, 256 threads.
// ---------------------------------------------------------------------------
__global__ __launch_bounds__(256) void latent_kernel(
    const float* __restrict__ h, const float* __restrict__ h0,
    const float* __restrict__ Wl1, const float* __restrict__ bl1,
    const float* __restrict__ Wl2, const float* __restrict__ bl2,
    const float* __restrict__ masks, float* __restrict__ out)
{
    const int m = blockIdx.x;                 // 0..799
    const int t = threadIdx.x;                // 0..255
    __shared__ float cat[2 * H_];
    __shared__ float zs[F_];
    if (t < H_)            cat[t] = h[(size_t)m * H_ + t];
    else if (t < 2 * H_)   cat[t] = h0[(size_t)m * H_ + (t - H_)];
    __syncthreads();

    float a = bl1[t];
    for (int k = 0; k < 2 * H_; ++k) a += Wl1[(size_t)t * (2 * H_) + k] * cat[k];
    zs[t] = sigmoidf_(a);
    __syncthreads();

    if (t < OUT_) {
        float o = bl2[t];
        for (int f = 0; f < F_; ++f) o += Wl2[(size_t)t * F_ + f] * zs[f];
        out[(size_t)m * OUT_ + t] = o * masks[m];
    }
}

// ---------------------------------------------------------------------------
extern "C" void kernel_launch(void* const* d_in, const int* in_sizes, int n_in,
                              void* d_out, int out_size, void* d_ws, size_t ws_size,
                              hipStream_t stream) {
    (void)in_sizes; (void)n_in; (void)out_size; (void)ws_size;

    const float* nodes = (const float*)d_in[0];   // [16,50,64]
    const float* edges = (const float*)d_in[1];   // [16,50,50,10]
    const float* masks = (const float*)d_in[2];   // [16,50,1]
    const float* We1   = (const float*)d_in[3];   // [128,10]
    const float* be1   = (const float*)d_in[4];   // [128]
    const float* We2   = (const float*)d_in[5];   // [4096,128]
    const float* be2   = (const float*)d_in[6];   // [4096]
    const float* W_ih  = (const float*)d_in[7];   // [192,64]
    const float* W_hh  = (const float*)d_in[8];   // [192,64]
    const float* b_ih  = (const float*)d_in[9];   // [192]
    const float* b_hh  = (const float*)d_in[10];  // [192]
    const float* Wl1   = (const float*)d_in[11];  // [256,128]
    const float* bl1   = (const float*)d_in[12];  // [256]
    const float* Wl2   = (const float*)d_in[13];  // [3,256]
    const float* bl2   = (const float*)d_in[14];  // [3]
    float* out = (float*)d_out;                   // [16,50,3]

    // Workspace layout (all offsets 256-aligned)
    char* ws = (char*)d_ws;
    unsigned short* e1T  = (unsigned short*)(ws);                 // 10,240,000 B
    unsigned short* we2p = (unsigned short*)(ws + 10240000);      //  1,048,576 B
    float* h_a  = (float*)(ws + 10240000 + 1048576);              //    204,800 B
    float* h_b  = (float*)(ws + 10240000 + 1048576 + 204800);     //    204,800 B
    float* msgs = (float*)(ws + 10240000 + 1048576 + 2 * 204800); //    204,800 B

    // Stage 1: edge layer-1 (sigmoid) -> bf16 transposed activations
    edge_l1_kernel<<<B_ * N_ * N_, 128, 0, stream>>>(edges, We1, be1, e1T);
    // Stage 2: pack We2 into WMMA-B fragment order (bf16, L2-resident 1 MB)
    pack_we2_kernel<<<1024, 32, 0, stream>>>(We2, we2p);
    // h0 = nodes_embed
    hipMemcpyAsync(h_a, nodes, (size_t)B_ * N_ * H_ * sizeof(float),
                   hipMemcpyDeviceToDevice, stream);

    float* cur = h_a;
    float* nxt = h_b;
    for (int step = 0; step < 3; ++step) {
        fused_msgs_kernel<<<B_ * N_, 256, 0, stream>>>(e1T, we2p, be2, cur, msgs);
        gru_kernel<<<B_ * N_, 64, 0, stream>>>(msgs, cur, W_ih, W_hh,
                                               b_ih, b_hh, masks, nxt);
        float* tmp = cur; cur = nxt; nxt = tmp;
    }

    latent_kernel<<<B_ * N_, 256, 0, stream>>>(cur, nodes, Wl1, bl1,
                                               Wl2, bl2, masks, out);
}
